// GNNMV_64175401337156
// MI455X (gfx1250) — matrix operand order repaired and checked by reading, not compiled, not run on hardware
//
#include <hip/hip_runtime.h>
#include <hip/hip_bf16.h>

// ---------------------------------------------------------------------------
// Types
// ---------------------------------------------------------------------------
typedef __attribute__((ext_vector_type(16))) __bf16          v16bf;
typedef __attribute__((ext_vector_type(8)))  float           v8f;
typedef __attribute__((ext_vector_type(8)))  unsigned short  ushort8;
typedef __attribute__((ext_vector_type(4)))  int             v4i;

// fp32 -> bf16 (round to nearest even), pure bit math
__device__ __forceinline__ unsigned short f2bf(float x) {
    unsigned int u = __float_as_uint(x);
    u += 0x7FFFu + ((u >> 16) & 1u);
    return (unsigned short)(u >> 16);
}

// Guaranteed hardware fp32 global atomic add (no-return form, STOREcnt-tracked).
__device__ __forceinline__ void atomicAddF32(float* p, float v) {
#if defined(__gfx1250__)
    asm volatile("global_atomic_add_f32 %0, %1, off" : : "v"(p), "v"(v) : "memory");
#else
    atomicAdd(p, v);
#endif
}

// 16-byte global -> LDS copy. Uses CDNA5 async-to-LDS when available
// (ASYNCcnt-tracked), otherwise a plain b128 load + ds_store.
// Probe result: param1 is `int4 __attribute__((address_space(1)))*` (global),
// 4 args total -> (global src, lds dst, imm offset, imm cpol).
#if defined(__gfx1250__) && __has_builtin(__builtin_amdgcn_global_load_async_to_lds_b128)
#define HAVE_ASYNC_LDS 1
#endif

__device__ __forceinline__ void copy16_g2l(const unsigned short* g, unsigned short* l) {
#if defined(HAVE_ASYNC_LDS)
    __builtin_amdgcn_global_load_async_to_lds_b128(
        (__attribute__((address_space(1))) v4i*)const_cast<unsigned short*>(g),
        (__attribute__((address_space(3))) v4i*)l, 0, 0);
#else
    *(ushort8*)l = *(const ushort8*)g;
#endif
}
__device__ __forceinline__ void wait_async_copies() {
#if defined(HAVE_ASYNC_LDS)
    asm volatile("s_wait_asynccnt 0x0" ::: "memory");
#endif
}

// ---------------------------------------------------------------------------
// WMMA GEMM: C[M x Nw] = A[M x K] * Wt[Nw x K]^T      (bf16 in, fp32 out)
//   A  : bf16 [M x K] row-major
//   Wt : bf16 [NwPad x K] (weights pre-transposed; NwPad = ceil64(Nw) rows valid)
// Block: 256 threads (8 waves), tile 128(M) x 64(N), BK=32.
// Wave grid 4(M) x 2(N); each wave computes 32x32 = 2x2 fragments of 16x16.
// Edge handling: source row indices are clamped (garbage only lands in output
// rows/cols that the guarded writeout discards).
// ---------------------------------------------------------------------------
__global__ __launch_bounds__(256) void gcn_wmma_gemm(
    const unsigned short* __restrict__ A, const unsigned short* __restrict__ Wt,
    float* __restrict__ C, int M, int K, int Nw)
{
    constexpr int BM = 128, BN = 64, BK = 32;
    constexpr int LDA_S = BK + 8;   // 40 halves = 80B rows: 16B aligned, conflict-free
    constexpr int LDB_S = BK + 8;
    __shared__ unsigned short sA[BM * LDA_S];   // [row][k]
    __shared__ unsigned short sB[BN * LDB_S];   // [col][k]

    const int t    = threadIdx.x;
    const int lane = t & 31;
    const int wid  = t >> 5;
    const int wm   = wid & 3;       // wave M index 0..3
    const int wn   = wid >> 2;      // wave N index 0..1
    const int blockM = blockIdx.x * BM;
    const int nBase  = blockIdx.y * BN;

    // ---- k-invariant staging coordinates (all 16B units, no guards needed)
    // A tile: 128x32 halves = 512 units; 2 units/thread
    const unsigned short* aSrc[2];
    unsigned short*       aDst[2];
    #pragma unroll
    for (int u = 0; u < 2; ++u) {
        int idx8 = t + u * 256;            // unit index
        int row  = idx8 >> 2;              // 0..127
        int col8 = (idx8 & 3) << 3;        // 0,8,16,24
        int gr   = blockM + row;
        int grc  = gr < M ? gr : (M - 1);  // clamp (safe: see header comment)
        aSrc[u] = A + (size_t)grc * K + col8;
        aDst[u] = &sA[row * LDA_S + col8];
    }
    // B tile: 64x32 halves = 256 units; 1 unit/thread
    const int bCol  = t >> 2;              // 0..63
    const int bPart = (t & 3) << 3;        // 0,8,16,24
    const unsigned short* bSrc = Wt + (size_t)(nBase + bCol) * K + bPart;
    unsigned short*       bDst = &sB[bCol * LDB_S + bPart];

    v8f acc[2][2] = {};

    for (int k0 = 0; k0 < K; k0 += BK) {
        copy16_g2l(aSrc[0] + k0, aDst[0]);
        copy16_g2l(aSrc[1] + k0, aDst[1]);
        copy16_g2l(bSrc   + k0, bDst);
        wait_async_copies();
        __syncthreads();

        // ---- build fragments per ISA 7.12.2 lane layouts
        union Frag { ushort8 h[2]; v16bf b; };
        Frag fa[2], fb[2];
        const int khA = (lane >> 4) * 8;       // A: lanes<16 -> K{0..7,16..23}
        const int khB = (lane >> 4) * 8;       // B: lanes<16 -> K{0..7,16..23}
        const int lr  = lane & 15;
        #pragma unroll
        for (int mi = 0; mi < 2; ++mi) {
            int row = wm * 32 + mi * 16 + lr;
            fa[mi].h[0] = *(const ushort8*)&sA[row * LDA_S + khA];
            fa[mi].h[1] = *(const ushort8*)&sA[row * LDA_S + khA + 16];
        }
        #pragma unroll
        for (int ni = 0; ni < 2; ++ni) {
            int col = wn * 32 + ni * 16 + lr;
            fb[ni].h[0] = *(const ushort8*)&sB[col * LDB_S + khB];
            fb[ni].h[1] = *(const ushort8*)&sB[col * LDB_S + khB + 16];
        }
        #pragma unroll
        for (int mi = 0; mi < 2; ++mi)
            #pragma unroll
            for (int ni = 0; ni < 2; ++ni)
                acc[mi][ni] = __builtin_amdgcn_wmma_f32_16x16x32_bf16(
                    false, fa[mi].b, false, fb[ni].b,
                    (short)0, acc[mi][ni], false, false);
        __syncthreads();
    }

    // ---- writeout: lane L, VGPR r -> M = r + 8*(L>>4), N = L&15
    const int rowHalf = (lane >> 4) * 8;
    const bool fullM  = (blockM + BM) <= M;
    #pragma unroll
    for (int mi = 0; mi < 2; ++mi)
        #pragma unroll
        for (int ni = 0; ni < 2; ++ni) {
            int gCol = nBase + wn * 32 + ni * 16 + (lane & 15);
            int row0 = blockM + wm * 32 + mi * 16 + rowHalf;
            float* p = C + (size_t)row0 * Nw + gCol;
            if (fullM && gCol < Nw) {
                #pragma unroll
                for (int r = 0; r < 8; ++r)
                    p[(size_t)r * Nw] = acc[mi][ni][r];
            } else {
                #pragma unroll
                for (int r = 0; r < 8; ++r)
                    if (row0 + r < M && gCol < Nw)
                        p[(size_t)r * Nw] = acc[mi][ni][r];
            }
        }
}

// ---------------------------------------------------------------------------
// Precision conversion kernels
// ---------------------------------------------------------------------------
__global__ void k_f32_to_bf16(const float* __restrict__ in, unsigned short* __restrict__ out,
                              int n) {
    int i = blockIdx.x * blockDim.x + threadIdx.x;
    if (i < n) out[i] = f2bf(in[i]);
}
// Wt[n][k] = bf16(W[k][n]); rows n >= Nw duplicate last column (clamped, finite).
__global__ void k_conv_w_t(const float* __restrict__ W, unsigned short* __restrict__ Wt,
                           int K, int Nw, int NwPad) {
    int i = blockIdx.x * blockDim.x + threadIdx.x;
    if (i < NwPad * K) {
        int n = i / K, k = i - n * K;
        int nc = n < Nw ? n : (Nw - 1);
        Wt[i] = f2bf(W[(size_t)k * Nw + nc]);
    }
}

// ---------------------------------------------------------------------------
// Degree / normalization helpers
// ---------------------------------------------------------------------------
__global__ void k_zero_u32(unsigned* p, int n) {
    int i = blockIdx.x * blockDim.x + threadIdx.x;
    if (i < n) p[i] = 0u;
}
__global__ void k_count_deg(const int* __restrict__ dst, unsigned* __restrict__ deg, int E) {
    int i = blockIdx.x * blockDim.x + threadIdx.x;
    if (i < E) atomicAdd(deg + dst[i], 1u);
}
__global__ void k_dinv(const unsigned* __restrict__ deg, float* __restrict__ dinv, int n) {
    int i = blockIdx.x * blockDim.x + threadIdx.x;
    if (i < n) dinv[i] = rsqrtf((float)deg[i] + 1.0f);   // +1 self loop; always > 0
}

// agg[n][f] = h[n][f] * dinv[n]^2   (self-loop term folded into init / zeroing)
__global__ void k_init_scaled(const float* __restrict__ h, const float* __restrict__ dinv,
                              float* __restrict__ agg, int total, int F) {
    int i = blockIdx.x * blockDim.x + threadIdx.x;
    if (i < total) {
        float s = dinv[i / F];
        agg[i] = h[i] * (s * s);
    }
}

// g[n][f] = bf16(relu(agg[n][f] + b[f]))  -- next layer consumes g only as GEMM A
__global__ void k_bias_relu_bf16(const float* __restrict__ agg, const float* __restrict__ b,
                                 unsigned short* __restrict__ g, int total, int F) {
    int i = blockIdx.x * blockDim.x + threadIdx.x;
    if (i < total) g[i] = f2bf(fmaxf(agg[i] + b[i % F], 0.f));
}

// out[n][f] = agg[n][f] + b[f]   (final layer, fp32, no relu; in-place OK)
__global__ void k_bias_f32(const float* __restrict__ agg, const float* __restrict__ b,
                           float* __restrict__ out, int total, int F) {
    int i = blockIdx.x * blockDim.x + threadIdx.x;
    if (i < total) out[i] = agg[i] + b[i % F];
}

// one wave per edge: agg[dst] += h[src] * dinv[src]*dinv[dst]
__global__ __launch_bounds__(256) void k_scatter(
    const float* __restrict__ h, const int* __restrict__ src, const int* __restrict__ dst,
    const float* __restrict__ dinv, float* __restrict__ agg, int E, int F)
{
    int e = blockIdx.x * 8 + (threadIdx.x >> 5);
    if (e >= E) return;
    int lane = threadIdx.x & 31;
    int s = src[e], d = dst[e];
    float w = dinv[s] * dinv[d];
    const float* hr = h + (size_t)s * F;
    float*       ar = agg + (size_t)d * F;
    if ((F & 127) == 0) {
        const float4* hr4 = (const float4*)hr;
        int nq = F >> 2;
        for (int j = lane; j < nq; j += 32) {
            float4 v = hr4[j];
            float* a = ar + j * 4;
            atomicAddF32(a + 0, v.x * w);
            atomicAddF32(a + 1, v.y * w);
            atomicAddF32(a + 2, v.z * w);
            atomicAddF32(a + 3, v.w * w);
        }
    } else {
        for (int f = lane; f < F; f += 32)
            atomicAddF32(ar + f, hr[f] * w);
    }
}

// ---------------------------------------------------------------------------
// Orchestration
// ---------------------------------------------------------------------------
extern "C" void kernel_launch(void* const* d_in, const int* in_sizes, int n_in,
                              void* d_out, int out_size, void* d_ws, size_t ws_size,
                              hipStream_t stream) {
    const float* x   = (const float*)d_in[0];
    const int*   ei  = (const int*)  d_in[1];
    const float* W1  = (const float*)d_in[3];
    const float* b1  = (const float*)d_in[4];
    const float* W2  = (const float*)d_in[5];
    const float* b2  = (const float*)d_in[6];
    const float* W3  = (const float*)d_in[7];
    const float* b3  = (const float*)d_in[8];
    float* out = (float*)d_out;

    const int N     = in_sizes[2];          // 50000
    const int E     = in_sizes[1] / 2;      // 800000
    const int F_IN  = in_sizes[0] / N;      // 512
    const int F_HID = in_sizes[4];          // 256
    const int F_OUT = in_sizes[8];          // 60
    const int F_OUT_PAD = (F_OUT + 63) & ~63;   // 64
    const int* src = ei;
    const int* dst = ei + E;

    // workspace layout (aligned 4KB chunks), ~155 MB total:
    //   deg | dinv | bufA(f32) | bufB(f32) | xb/gb(bf16) | w1t | w2t | w3t
    auto align4k = [](size_t v) { return (v + 4095) & ~(size_t)4095; };
    char* ws = (char*)d_ws;
    size_t o = 0;
    unsigned* deg  = (unsigned*)(ws + o);       o += align4k((size_t)N * 4);
    float*    dinv = (float*)(ws + o);          o += align4k((size_t)N * 4);
    float*    bufA = (float*)(ws + o);          o += align4k((size_t)N * F_HID * 4);
    float*    bufB = (float*)(ws + o);          o += align4k((size_t)N * F_HID * 4);
    unsigned short* xb  = (unsigned short*)(ws + o); o += align4k((size_t)N * F_IN * 2);
    unsigned short* gb  = xb;  // activations bf16; aliases xb (dead after GEMM1)
    unsigned short* w1t = (unsigned short*)(ws + o); o += align4k((size_t)F_HID * F_IN * 2);
    unsigned short* w2t = (unsigned short*)(ws + o); o += align4k((size_t)F_HID * F_HID * 2);
    unsigned short* w3t = (unsigned short*)(ws + o); o += align4k((size_t)F_OUT_PAD * F_HID * 2);

    const int TB = 256;
    auto cdiv = [](int a, int b) { return (a + b - 1) / b; };

    // ---- one-time conversions: x -> bf16, weights -> bf16 transposed
    k_f32_to_bf16<<<cdiv(N * F_IN, TB), TB, 0, stream>>>(x, xb, N * F_IN);
    k_conv_w_t<<<cdiv(F_HID * F_IN, TB), TB, 0, stream>>>(W1, w1t, F_IN, F_HID, F_HID);
    k_conv_w_t<<<cdiv(F_HID * F_HID, TB), TB, 0, stream>>>(W2, w2t, F_HID, F_HID, F_HID);
    k_conv_w_t<<<cdiv(F_OUT_PAD * F_HID, TB), TB, 0, stream>>>(W3, w3t, F_HID, F_OUT, F_OUT_PAD);

    // ---- degrees + D^{-1/2}
    k_zero_u32 <<<cdiv(N, TB), TB, 0, stream>>>(deg, N);
    k_count_deg<<<cdiv(E, TB), TB, 0, stream>>>(dst, deg, E);
    k_dinv     <<<cdiv(N, TB), TB, 0, stream>>>(deg, dinv, N);

    dim3 gHid(cdiv(N, 128), cdiv(F_HID, 64));   // (391, 4)
    dim3 gOut(cdiv(N, 128), cdiv(F_OUT, 64));   // (391, 1)
    const int totHid = N * F_HID;
    const int totOut = N * F_OUT;
    const int scatterBlocks = cdiv(E, 8);

    // ---- conv1: h1 = x @ W1 ; agg ; g1 = bf16(relu(agg + b1))
    gcn_wmma_gemm<<<gHid, TB, 0, stream>>>(xb, w1t, bufA, N, F_IN, F_HID);
    k_init_scaled  <<<cdiv(totHid, TB), TB, 0, stream>>>(bufA, dinv, bufB, totHid, F_HID);
    k_scatter      <<<scatterBlocks, TB, 0, stream>>>(bufA, src, dst, dinv, bufB, E, F_HID);
    k_bias_relu_bf16<<<cdiv(totHid, TB), TB, 0, stream>>>(bufB, b1, gb, totHid, F_HID);

    // ---- conv2: h2 = g1 @ W2 ; agg ; g2 = bf16(relu(agg + b2))  (mean-pool = identity)
    gcn_wmma_gemm<<<gHid, TB, 0, stream>>>(gb, w2t, bufA, N, F_HID, F_HID);
    k_init_scaled  <<<cdiv(totHid, TB), TB, 0, stream>>>(bufA, dinv, bufB, totHid, F_HID);
    k_scatter      <<<scatterBlocks, TB, 0, stream>>>(bufA, src, dst, dinv, bufB, E, F_HID);
    k_bias_relu_bf16<<<cdiv(totHid, TB), TB, 0, stream>>>(bufB, b2, gb, totHid, F_HID);

    // ---- conv3: h3 = g2 @ W3 ; agg directly into d_out ; +b3 (no relu)
    gcn_wmma_gemm<<<gOut, TB, 0, stream>>>(gb, w3t, bufA, N, F_HID, F_OUT);
    k_init_scaled<<<cdiv(totOut, TB), TB, 0, stream>>>(bufA, dinv, out, totOut, F_OUT);
    k_scatter    <<<scatterBlocks, TB, 0, stream>>>(bufA, src, dst, dinv, out, E, F_OUT);
    k_bias_f32   <<<cdiv(totOut, TB), TB, 0, stream>>>(out, b3, out, totOut, F_OUT);

    (void)n_in; (void)out_size; (void)ws_size;
}
